// AViTEncoder_60352880443886
// MI455X (gfx1250) — matrix operand order repaired and checked by reading, not compiled, not run on hardware
//
#include <hip/hip_runtime.h>
#include <cstdint>

typedef __bf16 bf16;
typedef __attribute__((ext_vector_type(16))) __bf16 v16bf;
typedef __attribute__((ext_vector_type(8)))  float  v8f;

#define BB   64
#define NTOK 197
#define DD   768
#define HH   12
#define FFD  3072
#define TOK  (BB*NTOK)   // 12608
#define BH   (BB*HH)     // 768
#define NPAD 224

// LDS staging geometry for the block-tiled GEMM
#define BM   128
#define BN   256
#define LDSROW 80                    // padded row stride (bytes) for 64B of k-data
#define A_BUF_BYTES (BM*LDSROW)      // 10240
#define B_BUF_BYTES (BN*LDSROW)      // 20480
#define SMEM_BYTES (2*A_BUF_BYTES + 2*B_BUF_BYTES)  // 61440

union Frag { v16bf v; uint4 q[2]; };

static __device__ __forceinline__ v8f wmma_bf16(v16bf a, v16bf b, v8f c) {
  return __builtin_amdgcn_wmma_f32_16x16x32_bf16(false, a, false, b, (short)0, c, false, false);
}

static __device__ __forceinline__ float gelu_tanh(float x) {
  float x3 = x * x * x;
  return 0.5f * x * (1.f + tanhf(0.79788456080286535588f * (x + 0.044715f * x3)));
}

static __device__ __forceinline__ void async_g2l_b128(uint32_t lds_addr, const void* gaddr) {
  asm volatile("global_load_async_to_lds_b128 %0, %1, off"
               :: "v"(lds_addr), "v"((uint64_t)(uintptr_t)gaddr) : "memory");
}
static __device__ __forceinline__ void wait_async0() {
  asm volatile("s_wait_asynccnt 0x0" ::: "memory");
}

// ---------------------------------------------------------------------------
// Halting mask: keep = newly_halted & retain<0.5; nearby_discard as gather.
// ---------------------------------------------------------------------------
__global__ __launch_bounds__(256) void mask_kernel(
    const float* __restrict__ x, const unsigned char* __restrict__ prev,
    const float* __restrict__ retain, float* __restrict__ active,
    float* __restrict__ maskOut)
{
  int idx = blockIdx.x * 256 + threadIdx.x;
  if (idx >= TOK) return;
  int t  = idx % NTOK;
  int b0 = idx - t;
  auto keepAt = [&](int tt) -> bool {
    float g0 = x[(size_t)(b0 + tt) * DD];           // sigmoid(g0*10-30)>0.5 <=> g0*10-30>0
    bool nh = (g0 * 10.f - 30.f > 0.f) && !prev[b0 + tt];
    return nh && (retain[b0 + tt] < 0.5f);
  };
  bool kp = keepAt(t);
  bool kill = false;
  if (t <= NTOK - 2 && (t % 14) != 0)       kill = kill || keepAt(t + 1); // left scatter
  if (t >= 1        && ((t + 1) % 14) != 0) kill = kill || keepAt(t - 1); // right scatter
  if (t == NTOK - 1)                        kill = kill || kp;            // clipped right edge
  bool m = !(prev[idx] || kp) && !kill;
  float mv = m ? 1.f : 0.f;
  active[idx]  = mv;
  maskOut[idx] = mv;
}

// ---------------------------------------------------------------------------
// LDS-tiled weight transpose + f32->bf16: Wt[n*K+k] = W[k*N+n].
// 64x64 tiles: coalesced f32 reads, coalesced 8B bf16 writes.
// K, N are multiples of 64 for all four weights.
// ---------------------------------------------------------------------------
__global__ __launch_bounds__(256) void transpose_w(
    const float* __restrict__ W, bf16* __restrict__ Wt, int K, int N)
{
  __shared__ bf16 tile[64][72];           // pad row to 144B (8B-aligned rows)
  int nBK = K >> 6;
  int bk = blockIdx.x % nBK, bn = blockIdx.x / nBK;
  int tk0 = bk << 6, tn0 = bn << 6;
  int tid = threadIdx.x;
#pragma unroll
  for (int it = 0; it < 16; ++it) {       // 4096 elements in
    int e = tid + it * 256;
    int row = e >> 6, col = e & 63;       // row = k index, col = n index
    tile[col][row] = (bf16)W[(size_t)(tk0 + row) * N + tn0 + col];
  }
  __syncthreads();
#pragma unroll
  for (int it = 0; it < 4; ++it) {        // 1024 chunks of 4 bf16 out
    int e = tid + it * 256;
    int n = e >> 4, c = e & 15;
    *(uint2*)&Wt[(size_t)(tn0 + n) * K + tk0 + c * 4] =
        *(const uint2*)&tile[n][c * 4];
  }
}

// ---------------------------------------------------------------------------
// LayerNorm: one wave per row of 768, bf16 out.
// ---------------------------------------------------------------------------
__global__ __launch_bounds__(256) void ln_kernel(
    const float* __restrict__ xin, const float* __restrict__ g,
    const float* __restrict__ bb, bf16* __restrict__ y, int rows)
{
  int gw   = (int)((blockIdx.x * (size_t)blockDim.x + threadIdx.x) >> 5);
  int lane = threadIdx.x & 31;
  if (gw >= rows) return;
  const float* xr = xin + (size_t)gw * DD;
  float vals[24];
  float s = 0.f, s2 = 0.f;
#pragma unroll
  for (int i = 0; i < 24; i++) {
    float v = xr[lane + i * 32];
    vals[i] = v; s += v; s2 += v * v;
  }
#pragma unroll
  for (int o = 16; o >= 1; o >>= 1) { s += __shfl_xor(s, o); s2 += __shfl_xor(s2, o); }
  float mu  = s * (1.f / DD);
  float var = s2 * (1.f / DD) - mu * mu;
  float rs  = rsqrtf(var + 1e-6f);
  bf16* yr = y + (size_t)gw * DD;
#pragma unroll
  for (int i = 0; i < 24; i++) {
    int c = lane + i * 32;
    yr[c] = (bf16)((vals[i] - mu) * rs * g[c] + bb[c]);
  }
}

// ---------------------------------------------------------------------------
// LDS-staged WMMA GEMM. Block tile 128x256, 8 waves (2x4), each wave 64x64.
// A/B k-slices (32 wide) staged via global_load_async_to_lds_b128, double
// buffered. stage() recomputes addresses from tid each call (rematerializable
// temps — keeping long-lived pointer IVs caused accumulator spills).
// Padded LDS rows (80B) give conflict-free ds_load_b128 fragments.
//  A: [M,K] bf16 row-major (lda=K stride), Bt: [N,K] bf16 (ldb=K stride).
//  mode 0: qkv scatter to [3][B,H,N,dh] bf16   (outB = qkv base)
//  mode 1: outF = resid + active*(acc+bias)    (f32, ld=DD)
//  mode 2: outB = bf16(gelu(acc+bias))         (ld=N)
//  mode 3: outF = outF + active*(acc+bias)     (f32, ld=DD, in-place residual)
// ---------------------------------------------------------------------------
__global__ __launch_bounds__(256) void gemm_staged_kernel(
    const bf16* __restrict__ A, int lda,
    const bf16* __restrict__ Bt, int ldb,
    const float* __restrict__ bias,
    int M, int N, int K, int mode,
    const float* __restrict__ resid,
    const float* __restrict__ act,
    float* __restrict__ outF,
    bf16*  __restrict__ outB)
{
  __shared__ uint4 smem4[SMEM_BYTES / 16];
  char* smem = (char*)smem4;
  const uint32_t ldsBase = (uint32_t)(uintptr_t)smem;   // low 32 bits = LDS offset

  int nBM = (M + BM - 1) / BM;
  int bm = blockIdx.x % nBM, bn = blockIdx.x / nBM;
  int blockM = bm * BM, blockN = bn * BN;

  int tid = threadIdx.x, lane = tid & 31, w = tid >> 5;
  int rlo = lane & 15, kg = lane >> 4;
  int mt = (w >> 2) * 64;      // wave row offset within block tile
  int nt = (w & 3) * 64;       // wave col offset within block tile

  // issue async copies of the 32-wide k-slice starting at k0 into buffer sel
  auto stage = [&](int k0, int sel) {
    uint32_t aDst = ldsBase + sel * A_BUF_BYTES;
    uint32_t bDst = ldsBase + 2 * A_BUF_BYTES + sel * B_BUF_BYTES;
    const bf16* Ak = A + k0;
    const bf16* Bk = Bt + k0;
#pragma unroll
    for (int it = 0; it < 2; ++it) {               // 512 chunks of 16B for A
      int e = tid + it * 256;
      int row = e >> 2, c = e & 3;
      int grow = blockM + row; if (grow >= M) grow = M - 1;
      async_g2l_b128(aDst + row * LDSROW + c * 16,
                     Ak + (size_t)grow * lda + c * 8);
    }
#pragma unroll
    for (int it = 0; it < 4; ++it) {               // 1024 chunks of 16B for B
      int e = tid + it * 256;
      int col = e >> 2, c = e & 3;
      async_g2l_b128(bDst + col * LDSROW + c * 16,
                     Bk + (size_t)(blockN + col) * ldb + c * 8);
    }
  };

  v8f zz = {0.f,0.f,0.f,0.f,0.f,0.f,0.f,0.f};
  v8f acc[4][4];
#pragma unroll
  for (int i = 0; i < 4; i++)
#pragma unroll
    for (int j = 0; j < 4; j++) acc[i][j] = zz;

  int nk = K >> 5;
  stage(0, 0);

#pragma unroll 1
  for (int kk = 0; kk < nk; ++kk) {
    wait_async0();
    __syncthreads();
    if (kk + 1 < nk) stage((kk + 1) << 5, (kk + 1) & 1);

    const char* aP = smem + (kk & 1) * A_BUF_BYTES;
    const char* bP = smem + 2 * A_BUF_BYTES + (kk & 1) * B_BUF_BYTES;
    Frag a[4], b[4];
#pragma unroll
    for (int i = 0; i < 4; i++) {
      const char* p = aP + (size_t)(mt + i * 16 + rlo) * LDSROW + kg * 16;
      a[i].q[0] = *(const uint4*)p;
      a[i].q[1] = *(const uint4*)(p + 32);
    }
#pragma unroll
    for (int j = 0; j < 4; j++) {
      const char* p = bP + (size_t)(nt + j * 16 + rlo) * LDSROW + kg * 16;
      b[j].q[0] = *(const uint4*)p;
      b[j].q[1] = *(const uint4*)(p + 32);
    }
#pragma unroll
    for (int i = 0; i < 4; i++)
#pragma unroll
      for (int j = 0; j < 4; j++)
        acc[i][j] = wmma_bf16(a[i].v, b[j].v, acc[i][j]);
  }

  int m0 = blockM + mt, n0 = blockN + nt;
#pragma unroll
  for (int i = 0; i < 4; i++) {
#pragma unroll
    for (int j = 0; j < 4; j++) {
      int col = n0 + j * 16 + rlo;
      float bv = bias[col];
#pragma unroll
      for (int r = 0; r < 8; r++) {
        int row = m0 + i * 16 + r + 8 * kg;
        if (row >= M) continue;
        float v = acc[i][j][r] + bv;
        if (mode == 0) {
          int which = col / DD;
          int hh = (col - which * DD) >> 6;
          int d  = col & 63;
          int bb = row / NTOK;
          int nn = row - bb * NTOK;
          outB[(size_t)which * BH * NTOK * 64 +
               (((size_t)(bb * HH + hh)) * NTOK + nn) * 64 + d] = (bf16)v;
        } else if (mode == 1) {
          size_t idx = (size_t)row * DD + col;
          outF[idx] = resid[idx] + act[row] * v;
        } else if (mode == 2) {
          outB[(size_t)row * N + col] = (bf16)gelu_tanh(v);
        } else {
          size_t idx = (size_t)row * DD + col;
          outF[idx] = outF[idx] + act[row] * v;
        }
      }
    }
  }
}

// ---------------------------------------------------------------------------
// Fused attention per (b,h): scores via WMMA, masked softmax, P*V via WMMA.
// ---------------------------------------------------------------------------
__global__ __launch_bounds__(128) void attn_kernel(
    const bf16* __restrict__ qkv, const float* __restrict__ act,
    bf16* __restrict__ obf)
{
  __shared__ bf16  vT[64][NPAD];        // V transposed (d-major), zero-padded
  __shared__ bf16  psm[4][16][NPAD];    // per-wave probability tile
  __shared__ float amask[NPAD];
  int bh = blockIdx.x;
  int b  = bh / HH;
  int h  = bh - b * HH;
  const size_t seg = (size_t)BH * NTOK * 64;
  const bf16* qp = qkv + (size_t)bh * NTOK * 64;
  const bf16* kp = qp + seg;
  const bf16* vp = qp + 2 * seg;
  int tid = threadIdx.x, lane = tid & 31, w = tid >> 5;
  int rlo = lane & 15, kg = lane >> 4;

  for (int i = tid; i < 64 * NPAD; i += 128) {
    int d = i & 63, j = i >> 6;
    vT[d][j] = (j < NTOK) ? vp[(size_t)j * 64 + d] : (bf16)0.f;
  }
  for (int j = tid; j < NPAD; j += 128)
    amask[j] = (j < NTOK) ? act[b * NTOK + j] : 0.f;
  __syncthreads();

  for (int t = w; t < 13; t += 4) {
    int m0 = t * 16;
    int qrow = m0 + rlo; if (qrow > NTOK - 1) qrow = NTOK - 1;
    Frag aq0, aq1;
    {
      const bf16* p = qp + (size_t)qrow * 64 + kg * 8;
      aq0.q[0] = *(const uint4*)p;        aq0.q[1] = *(const uint4*)(p + 16);
      aq1.q[0] = *(const uint4*)(p + 32); aq1.q[1] = *(const uint4*)(p + 48);
    }
    v8f s[13];
#pragma unroll
    for (int j = 0; j < 13; j++) {
      int krow = j * 16 + rlo; if (krow > NTOK - 1) krow = NTOK - 1;
      const bf16* p = kp + (size_t)krow * 64 + kg * 8;
      Frag b0, b1;
      b0.q[0] = *(const uint4*)p;        b0.q[1] = *(const uint4*)(p + 16);
      b1.q[0] = *(const uint4*)(p + 32); b1.q[1] = *(const uint4*)(p + 48);
      v8f z = {0.f,0.f,0.f,0.f,0.f,0.f,0.f,0.f};
      z = wmma_bf16(aq0.v, b0.v, z);
      z = wmma_bf16(aq1.v, b1.v, z);
      s[j] = z;
    }
    // mask + softmax (per row r, reduce across 16 lanes of the half-wave)
    float mx[8];
#pragma unroll
    for (int r = 0; r < 8; r++) mx[r] = -3.0e38f;
#pragma unroll
    for (int j = 0; j < 13; j++) {
      int col = j * 16 + rlo;
      bool pad = (col >= NTOK);
      bool on  = !pad && (amask[col] > 0.5f);
#pragma unroll
      for (int r = 0; r < 8; r++) {
        float v = on ? s[j][r] * 0.125f : (pad ? -__builtin_inff() : -1.0e9f);
        s[j][r] = v;
        mx[r] = fmaxf(mx[r], v);
      }
    }
#pragma unroll
    for (int o = 8; o >= 1; o >>= 1)
#pragma unroll
      for (int r = 0; r < 8; r++) mx[r] = fmaxf(mx[r], __shfl_xor(mx[r], o));
    float sum[8];
#pragma unroll
    for (int r = 0; r < 8; r++) sum[r] = 0.f;
#pragma unroll
    for (int j = 0; j < 13; j++)
#pragma unroll
      for (int r = 0; r < 8; r++) {
        float e = __expf(s[j][r] - mx[r]);
        s[j][r] = e; sum[r] += e;
      }
#pragma unroll
    for (int o = 8; o >= 1; o >>= 1)
#pragma unroll
      for (int r = 0; r < 8; r++) sum[r] += __shfl_xor(sum[r], o);
    float inv[8];
#pragma unroll
    for (int r = 0; r < 8; r++) inv[r] = 1.f / sum[r];
    // C-layout -> A-layout via LDS tile
#pragma unroll
    for (int j = 0; j < 13; j++) {
      int col = j * 16 + rlo;
#pragma unroll
      for (int r = 0; r < 8; r++)
        psm[w][r + 8 * kg][col] = (bf16)(s[j][r] * inv[r]);
    }
#pragma unroll
    for (int r = 0; r < 8; r++) psm[w][r + 8 * kg][208 + rlo] = (bf16)0.f;

    // P @ V
#pragma unroll
    for (int jd = 0; jd < 4; jd++) {
      v8f o8 = {0.f,0.f,0.f,0.f,0.f,0.f,0.f,0.f};
#pragma unroll
      for (int k0 = 0; k0 < NPAD; k0 += 32) {
        Frag pa, vb;
        const bf16* pp = &psm[w][rlo][k0 + kg * 8];
        pa.q[0] = *(const uint4*)pp; pa.q[1] = *(const uint4*)(pp + 16);
        const bf16* vv = &vT[jd * 16 + rlo][k0 + kg * 8];
        vb.q[0] = *(const uint4*)vv; vb.q[1] = *(const uint4*)(vv + 16);
        o8 = wmma_bf16(pa.v, vb.v, o8);
      }
#pragma unroll
      for (int r = 0; r < 8; r++) {
        int row = m0 + r + 8 * kg;
        if (row < NTOK)
          obf[((size_t)b * NTOK + row) * DD + h * 64 + jd * 16 + rlo] = (bf16)o8[r];
      }
    }
  }
}

// ---------------------------------------------------------------------------
extern "C" void kernel_launch(void* const* d_in, const int* in_sizes, int n_in,
                              void* d_out, int out_size, void* d_ws, size_t ws_size,
                              hipStream_t stream) {
  (void)in_sizes; (void)n_in; (void)out_size; (void)ws_size;
  const float* x            = (const float*)d_in[0];
  const unsigned char* prev = (const unsigned char*)d_in[1];
  const float* retain       = (const float*)d_in[2];
  const float* ln1g = (const float*)d_in[3];
  const float* ln1b = (const float*)d_in[4];
  const float* ln2g = (const float*)d_in[5];
  const float* ln2b = (const float*)d_in[6];
  const float* Wqkv = (const float*)d_in[7];
  const float* bqkv = (const float*)d_in[8];
  const float* Wo   = (const float*)d_in[9];
  const float* bo   = (const float*)d_in[10];
  const float* W1   = (const float*)d_in[11];
  const float* b1   = (const float*)d_in[12];
  const float* W2   = (const float*)d_in[13];
  const float* b2   = (const float*)d_in[14];

  float* outX    = (float*)d_out;            // [TOK, DD] final x
  float* outMask = outX + (size_t)TOK * DD;  // [TOK] mask_token as float

  char* base = (char*)d_ws;
  size_t off = 0;
  auto alloc = [&](size_t bytes) -> char* {
    char* p = base + off;
    off += (bytes + 255) & ~(size_t)255;
    return p;
  };
  float* active = (float*)alloc((size_t)TOK * 4);
  bf16* ybf   = (bf16*)alloc((size_t)TOK * DD * 2);       // LN1 out; reused for LN2 out
  bf16* WqkvT = (bf16*)alloc((size_t)DD * 3 * DD * 2);
  bf16* WoT   = (bf16*)alloc((size_t)DD * DD * 2);
  bf16* W1T   = (bf16*)alloc((size_t)DD * FFD * 2);
  bf16* W2T   = (bf16*)alloc((size_t)FFD * DD * 2);
  bf16* qkvbf = (bf16*)alloc((size_t)3 * BH * NTOK * 64 * 2); // 58,097,664 B (256-aligned)
  bf16* obf   = (bf16*)alloc((size_t)TOK * DD * 2);           // contiguous after qkvbf
  bf16* hbf   = qkvbf;   // qkvbf+obf region == exactly TOK*FFD bf16 for GELU intermediate

  // weight transpose/convert (64x64 LDS tiles)
  auto trBlocks = [](int K, int N) { return (unsigned)((K >> 6) * (N >> 6)); };
  transpose_w<<<trBlocks(DD, 3 * DD), 256, 0, stream>>>(Wqkv, WqkvT, DD, 3 * DD);
  transpose_w<<<trBlocks(DD, DD), 256, 0, stream>>>(Wo, WoT, DD, DD);
  transpose_w<<<trBlocks(DD, FFD), 256, 0, stream>>>(W1, W1T, DD, FFD);
  transpose_w<<<trBlocks(FFD, DD), 256, 0, stream>>>(W2, W2T, FFD, DD);

  mask_kernel<<<(TOK + 255) / 256, 256, 0, stream>>>(x, prev, retain, active, outMask);

  // LN1 -> ybf
  ln_kernel<<<(TOK * 32 + 255) / 256, 256, 0, stream>>>(x, ln1g, ln1b, ybf, TOK);

  auto gemmBlocks = [](int M, int N) {
    return (unsigned)(((M + BM - 1) / BM) * (N / BN));
  };

  // QKV: [TOK,768] @ [768,2304] -> scatter bf16 q/k/v
  gemm_staged_kernel<<<gemmBlocks(TOK, 3 * DD), 256, 0, stream>>>(
      ybf, DD, WqkvT, DD, bqkv, TOK, 3 * DD, DD, 0, nullptr, nullptr, nullptr, qkvbf);

  // attention per (b,h)
  attn_kernel<<<BH, 128, 0, stream>>>(qkvbf, active, obf);

  // x1 = x + active*(o @ Wo + bo) -> d_out
  gemm_staged_kernel<<<gemmBlocks(TOK, DD), 256, 0, stream>>>(
      obf, DD, WoT, DD, bo, TOK, DD, DD, 1, x, active, outX, nullptr);

  // LN2 on x1 -> ybf (reuse)
  ln_kernel<<<(TOK * 32 + 255) / 256, 256, 0, stream>>>(outX, ln2g, ln2b, ybf, TOK);

  // h = gelu(y2 @ W1 + b1) bf16 (reuses qkv+obf region)
  gemm_staged_kernel<<<gemmBlocks(TOK, FFD), 256, 0, stream>>>(
      ybf, DD, W1T, DD, b1, TOK, FFD, DD, 2, nullptr, nullptr, nullptr, hbf);

  // x = x1 + active*(h @ W2 + b2) in-place on d_out
  gemm_staged_kernel<<<gemmBlocks(TOK, DD), 256, 0, stream>>>(
      hbf, FFD, W2T, FFD, b2, TOK, DD, FFD, 3, nullptr, active, outX, nullptr);
}
